// BiHgru1d_60997125538133
// MI455X (gfx1250) — matrix operand order repaired and checked by reading, not compiled, not run on hardware
//
#include <hip/hip_runtime.h>

// Problem constants (reference: n=8192, b=4, d=1024)
#define SEQ_N   8192
#define BATCH   4
#define DIM     1024
#define TWO_D   2048
#define TOKENS  (SEQ_N * BATCH)          // 32768 rows
#define KIN     DIM                      // GEMM1 K
#define NCAT    (2 * TWO_D + DIM)        // 5120 fused output cols: [Wi | Wg | Wl]
#define NCHUNK  128
#define CHUNK_L (SEQ_N / NCHUNK)         // 64
#define BD      (BATCH * DIM)            // 4096 independent scan channels

typedef __attribute__((ext_vector_type(16))) __bf16 v16bf;
typedef __attribute__((ext_vector_type(8)))  __bf16 v8bf;
typedef __attribute__((ext_vector_type(8)))  float  v8f;

__device__ __forceinline__ unsigned short f2bf(float x) {
  unsigned int u = __float_as_uint(x);
  u += 0x7FFFu + ((u >> 16) & 1u);       // round-to-nearest-even
  return (unsigned short)(u >> 16);
}
__device__ __forceinline__ float silu_f(float x) {
  return x * (1.0f / (1.0f + __expf(-x)));
}
__device__ __forceinline__ float sigmoid_f(float x) {
  return 1.0f / (1.0f + __expf(-x));
}

// ---- WMMA fragment loads (16-bit A / B, same contiguous k order k = 16*(lane/16) + s) ----
// A (16x32, row-major [M][K]): lane<16 holds M=lane, K=k0+0..15;
// lane>=16 holds M=lane-16, K=k0+16..31  -> two 16B loads.
__device__ __forceinline__ v16bf load_frag_a(const unsigned short* __restrict__ A,
                                             int row0, int ld, int k0) {
  int lane = threadIdx.x & 31;
  const unsigned short* p =
      A + (size_t)(row0 + (lane & 15)) * ld + k0 + ((lane >> 4) << 4);
  union { v16bf v; v8bf h[2]; } u;
  u.h[0] = *reinterpret_cast<const v8bf*>(p);       // K + {0..7} / {16..23}
  u.h[1] = *reinterpret_cast<const v8bf*>(p + 8);   // K + {8..15} / {24..31}
  return u.v;
}
// B (32x16) loaded from pre-transposed W^T [N][K]: lane<16 -> N=lane, K=k0..k0+15;
// lane>=16 -> N=lane-16, K=k0+16..k0+31  -> one 32B load.
__device__ __forceinline__ v16bf load_frag_b(const unsigned short* __restrict__ BT,
                                             int col0, int ld, int k0) {
  int lane = threadIdx.x & 31;
  const unsigned short* p =
      BT + (size_t)(col0 + (lane & 15)) * ld + k0 + ((lane >> 4) << 4);
  return *reinterpret_cast<const v16bf*>(p);
}

// ---------------- prep kernels ----------------
__global__ void k_convert_bf16(const float* __restrict__ src,
                               unsigned short* __restrict__ dst, size_t total) {
  size_t i = (size_t)blockIdx.x * blockDim.x + threadIdx.x;
  if (i < total) dst[i] = f2bf(src[i]);
}

// W [K][N] row-major -> WT [N][K] bf16
__global__ void k_transpose_bf16(const float* __restrict__ W,
                                 unsigned short* __restrict__ WT,
                                 int Kd, int Nd) {
  size_t i = (size_t)blockIdx.x * blockDim.x + threadIdx.x;
  size_t total = (size_t)Kd * Nd;
  if (i >= total) return;
  int k = (int)(i / Nd);
  int n = (int)(i % Nd);
  WT[(size_t)n * Kd + k] = f2bf(W[i]);
}

__global__ void k_theta_prep(const float* __restrict__ thf,
                             const float* __restrict__ thr,
                             float* __restrict__ trig) {  // [4][DIM]: cF,sF,cR,sR
  int c = blockIdx.x * blockDim.x + threadIdx.x;
  if (c < DIM) {
    trig[c]           = __cosf(thf[c]);
    trig[DIM + c]     = __sinf(thf[c]);
    trig[2 * DIM + c] = __cosf(thr[c]);
    trig[3 * DIM + c] = __sinf(thr[c]);
  }
}

// ---------------- GEMM1: Y = x @ [Wi|Wg|Wl] + bias, fused activations ----------------
// 256 threads = 8 waves; wave tile 32x32 (2x2 WMMA frags); block tile 64(M) x 128(N).
__global__ __launch_bounds__(256)
void k_gemm1_wmma(const unsigned short* __restrict__ Abf,   // [TOKENS][KIN] bf16
                  const unsigned short* __restrict__ WcatT, // [NCAT][KIN]  bf16
                  const float* __restrict__ bi, const float* __restrict__ bg,
                  const float* __restrict__ bl,
                  float* __restrict__ ir, float* __restrict__ ii,
                  float* __restrict__ gate, float* __restrict__ lam) {
  const int K = KIN;
  int lane = threadIdx.x & 31;
  int wave = threadIdx.x >> 5;
  int row0 = blockIdx.y * 64 + (wave & 1) * 32;
  int col0 = blockIdx.x * 128 + (wave >> 1) * 32;

  v8f acc00 = {}, acc01 = {}, acc10 = {}, acc11 = {};
  for (int k0 = 0; k0 < K; k0 += 32) {
    if (k0 + 64 < K)
      __builtin_prefetch(Abf + (size_t)row0 * K + k0 + 64, 0, 0);
    v16bf a0 = load_frag_a(Abf, row0,      K, k0);
    v16bf a1 = load_frag_a(Abf, row0 + 16, K, k0);
    v16bf b0 = load_frag_b(WcatT, col0,      K, k0);
    v16bf b1 = load_frag_b(WcatT, col0 + 16, K, k0);
    acc00 = __builtin_amdgcn_wmma_f32_16x16x32_bf16(false, a0, false, b0, (short)0, acc00, false, false);
    acc01 = __builtin_amdgcn_wmma_f32_16x16x32_bf16(false, a0, false, b1, (short)0, acc01, false, false);
    acc10 = __builtin_amdgcn_wmma_f32_16x16x32_bf16(false, a1, false, b0, (short)0, acc10, false, false);
    acc11 = __builtin_amdgcn_wmma_f32_16x16x32_bf16(false, a1, false, b1, (short)0, acc11, false, false);
  }

  // Epilogue: C/D layout -> lane: N = lane&15; VGPR v: M = v + 8*(lane>=16)
  int laneN = lane & 15;
  int moff  = (lane >> 4) << 3;
#pragma unroll
  for (int mi = 0; mi < 2; ++mi) {
#pragma unroll
    for (int ni = 0; ni < 2; ++ni) {
      v8f acc = (mi == 0) ? (ni == 0 ? acc00 : acc01) : (ni == 0 ? acc10 : acc11);
      int c = col0 + ni * 16 + laneN;
      int rbase = row0 + mi * 16 + moff;
      float bias; int region;
      if (c < TWO_D)            { bias = bi[c];             region = 0; }
      else if (c < 2 * TWO_D)   { bias = bg[c - TWO_D];     region = 1; }
      else                      { bias = bl[c - 2 * TWO_D]; region = 2; }
#pragma unroll
      for (int v = 0; v < 8; ++v) {
        int r = rbase + v;
        float val = acc[v] + bias;
        if (region == 0) {        // input_state -> deinterleave real/imag (silu, unscaled)
          float s = silu_f(val);
          float* dst = (c & 1) ? ii : ir;
          dst[(size_t)r * DIM + (c >> 1)] = s;
        } else if (region == 1) { // gate = silu
          gate[(size_t)r * TWO_D + (c - TWO_D)] = silu_f(val);
        } else {                  // lam = sigmoid (lower_bound = 0)
          lam[(size_t)r * DIM + (c - 2 * TWO_D)] = sigmoid_f(val);
        }
      }
    }
  }
}

// ---------------- u scaling: u *= (1 - lam) ----------------
__global__ void k_scale_u(float* __restrict__ ir, float* __restrict__ ii,
                          const float* __restrict__ lam, size_t total) {
  size_t i = (size_t)blockIdx.x * blockDim.x + threadIdx.x;
  if (i < total) {
    float om = 1.0f - lam[i];
    ir[i] *= om;
    ii[i] *= om;
  }
}

// ---------------- chunked complex scan ----------------
// pass1: per-(chunk,channel) compute chunk-local (G = prod g, H = local scan end)
__global__ __launch_bounds__(256)
void k_scan_pass1(const float* __restrict__ lam, const float* __restrict__ ir,
                  const float* __restrict__ ii,
                  const float* __restrict__ cosA, const float* __restrict__ sinA,
                  float* __restrict__ Gr, float* __restrict__ Gi,
                  float* __restrict__ Hr, float* __restrict__ Hi, int dir) {
  int tid = blockIdx.x * blockDim.x + threadIdx.x;  // NCHUNK*BD threads
  int ch = tid & (BD - 1);
  int chunk = tid / BD;
  int c = ch & (DIM - 1);
  float cA = cosA[c], sA = sinA[c];
  float gr = 1.0f, gi = 0.0f, hr = 0.0f, hi = 0.0f;
  int jbase = chunk * CHUNK_L;
  for (int j = 0; j < CHUNK_L; ++j) {
    int jj = jbase + j;
    int pos = dir ? (SEQ_N - 1 - jj) : jj;
    size_t e = (size_t)pos * BD + ch;
    float la = lam[e];
    float tgr = la * cA, tgi = la * sA;
    float nhr = tgr * hr - tgi * hi + ir[e];
    float nhi = tgr * hi + tgi * hr + ii[e];
    float ngr = tgr * gr - tgi * gi;
    float ngi = tgr * gi + tgi * gr;
    hr = nhr; hi = nhi; gr = ngr; gi = ngi;
  }
  size_t o = (size_t)chunk * BD + ch;
  Gr[o] = gr; Gi[o] = gi; Hr[o] = hr; Hi[o] = hi;
}

// pass2: exclusive inter-chunk scan; overwrites (Hr,Hi) with carry-in per chunk
__global__ void k_scan_pass2(float* __restrict__ Gr, float* __restrict__ Gi,
                             float* __restrict__ Hr, float* __restrict__ Hi) {
  int ch = blockIdx.x * blockDim.x + threadIdx.x;
  if (ch >= BD) return;
  float cr = 0.0f, ci = 0.0f;
  for (int chunk = 0; chunk < NCHUNK; ++chunk) {
    size_t o = (size_t)chunk * BD + ch;
    float gr = Gr[o], gi = Gi[o], hr = Hr[o], hi = Hi[o];
    Hr[o] = cr; Hi[o] = ci;
    float ncr = gr * cr - gi * ci + hr;
    float nci = gr * ci + gi * cr + hi;
    cr = ncr; ci = nci;
  }
}

// pass3: replay chunk from carry-in, write (fwd) or accumulate (rev) into feature
__global__ __launch_bounds__(256)
void k_scan_pass3(const float* __restrict__ lam, const float* __restrict__ ir,
                  const float* __restrict__ ii,
                  const float* __restrict__ cosA, const float* __restrict__ sinA,
                  const float* __restrict__ Hr, const float* __restrict__ Hi,
                  float* __restrict__ feature, int dir, int accumulate) {
  int tid = blockIdx.x * blockDim.x + threadIdx.x;
  int ch = tid & (BD - 1);
  int chunk = tid / BD;
  int c = ch & (DIM - 1);
  int ib = ch / DIM;
  float cA = cosA[c], sA = sinA[c];
  size_t o = (size_t)chunk * BD + ch;
  float hr = Hr[o], hi = Hi[o];
  int jbase = chunk * CHUNK_L;
  for (int j = 0; j < CHUNK_L; ++j) {
    int jj = jbase + j;
    int pos = dir ? (SEQ_N - 1 - jj) : jj;
    size_t e = (size_t)pos * BD + ch;
    float la = lam[e];
    float tgr = la * cA, tgi = la * sA;
    float nhr = tgr * hr - tgi * hi + ir[e];
    float nhi = tgr * hi + tgi * hr + ii[e];
    hr = nhr; hi = nhi;
    size_t frow = (size_t)(pos * BATCH + ib) * TWO_D;
    if (accumulate) {
      feature[frow + c]       += hr;
      feature[frow + DIM + c] += hi;
    } else {
      feature[frow + c]       = hr;
      feature[frow + DIM + c] = hi;
    }
  }
}

// ---------------- LayerNorm * gate -> bf16 GEMM2 input ----------------
__global__ __launch_bounds__(256)
void k_ln_gate(const float* __restrict__ feature, const float* __restrict__ gate,
               const float* __restrict__ lng, const float* __restrict__ lnb,
               unsigned short* __restrict__ gbf) {
  __shared__ float s1[256];
  __shared__ float s2[256];
  int r = blockIdx.x;
  int t = threadIdx.x;
  const float* row = feature + (size_t)r * TWO_D;
  float sum = 0.0f, sq = 0.0f;
  for (int c = t; c < TWO_D; c += 256) {
    float v = row[c];
    sum += v; sq += v * v;
  }
  s1[t] = sum; s2[t] = sq;
  __syncthreads();
  for (int off = 128; off > 0; off >>= 1) {
    if (t < off) { s1[t] += s1[t + off]; s2[t] += s2[t + off]; }
    __syncthreads();
  }
  float mu  = s1[0] * (1.0f / TWO_D);
  float var = s2[0] * (1.0f / TWO_D) - mu * mu;
  float inv = rsqrtf(var + 1e-5f);
  for (int c = t; c < TWO_D; c += 256) {
    float v = (row[c] - mu) * inv * lng[c] + lnb[c];
    gbf[(size_t)r * TWO_D + c] = f2bf(v * gate[(size_t)r * TWO_D + c]);
  }
}

// ---------------- GEMM2: out = (feat*gate) @ Wo + bo ----------------
__global__ __launch_bounds__(256)
void k_gemm2_wmma(const unsigned short* __restrict__ Abf,  // [TOKENS][TWO_D] bf16
                  const unsigned short* __restrict__ WoT,  // [DIM][TWO_D]    bf16
                  const float* __restrict__ bo,
                  float* __restrict__ out) {
  const int K = TWO_D;
  int lane = threadIdx.x & 31;
  int wave = threadIdx.x >> 5;
  int row0 = blockIdx.y * 64 + (wave & 1) * 32;
  int col0 = blockIdx.x * 128 + (wave >> 1) * 32;

  v8f acc00 = {}, acc01 = {}, acc10 = {}, acc11 = {};
  for (int k0 = 0; k0 < K; k0 += 32) {
    if (k0 + 64 < K)
      __builtin_prefetch(Abf + (size_t)row0 * K + k0 + 64, 0, 0);
    v16bf a0 = load_frag_a(Abf, row0,      K, k0);
    v16bf a1 = load_frag_a(Abf, row0 + 16, K, k0);
    v16bf b0 = load_frag_b(WoT, col0,      K, k0);
    v16bf b1 = load_frag_b(WoT, col0 + 16, K, k0);
    acc00 = __builtin_amdgcn_wmma_f32_16x16x32_bf16(false, a0, false, b0, (short)0, acc00, false, false);
    acc01 = __builtin_amdgcn_wmma_f32_16x16x32_bf16(false, a0, false, b1, (short)0, acc01, false, false);
    acc10 = __builtin_amdgcn_wmma_f32_16x16x32_bf16(false, a1, false, b0, (short)0, acc10, false, false);
    acc11 = __builtin_amdgcn_wmma_f32_16x16x32_bf16(false, a1, false, b1, (short)0, acc11, false, false);
  }

  int laneN = lane & 15;
  int moff  = (lane >> 4) << 3;
#pragma unroll
  for (int mi = 0; mi < 2; ++mi) {
#pragma unroll
    for (int ni = 0; ni < 2; ++ni) {
      v8f acc = (mi == 0) ? (ni == 0 ? acc00 : acc01) : (ni == 0 ? acc10 : acc11);
      int c = col0 + ni * 16 + laneN;
      int rbase = row0 + mi * 16 + moff;
      float bias = bo[c];
#pragma unroll
      for (int v = 0; v < 8; ++v) {
        out[(size_t)(rbase + v) * DIM + c] = acc[v] + bias;
      }
    }
  }
}

extern "C" void kernel_launch(void* const* d_in, const int* in_sizes, int n_in,
                              void* d_out, int out_size, void* d_ws, size_t ws_size,
                              hipStream_t stream) {
  (void)in_sizes; (void)n_in; (void)out_size; (void)ws_size;
  const float* x   = (const float*)d_in[0];
  const float* Wi  = (const float*)d_in[1];
  const float* bi  = (const float*)d_in[2];
  const float* Wl  = (const float*)d_in[3];
  const float* bl  = (const float*)d_in[4];
  const float* thf = (const float*)d_in[5];
  const float* thr = (const float*)d_in[6];
  const float* Wg  = (const float*)d_in[7];
  const float* bg  = (const float*)d_in[8];
  const float* Wo  = (const float*)d_in[9];
  const float* bo  = (const float*)d_in[10];
  const float* lng = (const float*)d_in[11];
  const float* lnb = (const float*)d_in[12];

  char* ws = (char*)d_ws;
  size_t off = 0;
  auto alloc = [&](size_t bytes) -> char* {
    char* p = ws + off;
    off += (bytes + 255) & ~(size_t)255;
    return p;
  };
  const size_t TD  = (size_t)TOKENS * DIM;
  const size_t T2D = (size_t)TOKENS * TWO_D;

  float* irp   = (float*)alloc(TD * 4);
  float* iip   = (float*)alloc(TD * 4);
  float* lamp  = (float*)alloc(TD * 4);
  float* gatep = (float*)alloc(T2D * 4);
  float* featp = (float*)alloc(T2D * 4);
  unsigned short* xbf  = (unsigned short*)alloc(TD * 2);
  unsigned short* wcat = (unsigned short*)alloc((size_t)NCAT * KIN * 2);
  unsigned short* wot  = (unsigned short*)alloc((size_t)DIM * TWO_D * 2);
  float* trig = (float*)alloc(4 * DIM * 4);
  const size_t SC = (size_t)NCHUNK * BD;
  float* GrF = (float*)alloc(SC * 4);
  float* GiF = (float*)alloc(SC * 4);
  float* HrF = (float*)alloc(SC * 4);
  float* HiF = (float*)alloc(SC * 4);
  float* GrR = (float*)alloc(SC * 4);
  float* GiR = (float*)alloc(SC * 4);
  float* HrR = (float*)alloc(SC * 4);
  float* HiR = (float*)alloc(SC * 4);
  unsigned short* gbf = (unsigned short*)alloc(T2D * 2);

  dim3 b256(256);

  // prep: bf16 conversions / transposes / trig
  k_convert_bf16<<<(unsigned)((TD + 255) / 256), b256, 0, stream>>>(x, xbf, TD);
  k_transpose_bf16<<<(DIM * TWO_D + 255) / 256, b256, 0, stream>>>(Wi, wcat, DIM, TWO_D);
  k_transpose_bf16<<<(DIM * TWO_D + 255) / 256, b256, 0, stream>>>(
      Wg, wcat + (size_t)TWO_D * KIN, DIM, TWO_D);
  k_transpose_bf16<<<(DIM * DIM + 255) / 256, b256, 0, stream>>>(
      Wl, wcat + (size_t)2 * TWO_D * KIN, DIM, DIM);
  k_transpose_bf16<<<(TWO_D * DIM + 255) / 256, b256, 0, stream>>>(Wo, wot, TWO_D, DIM);
  k_theta_prep<<<(DIM + 255) / 256, b256, 0, stream>>>(thf, thr, trig);

  // fused input GEMM (x @ [Wi|Wg|Wl]) with silu/silu/sigmoid epilogues
  k_gemm1_wmma<<<dim3(NCAT / 128, TOKENS / 64), b256, 0, stream>>>(
      xbf, wcat, bi, bg, bl, irp, iip, gatep, lamp);

  // u = (1-lam) * silu(...)
  k_scale_u<<<(unsigned)((TD + 255) / 256), b256, 0, stream>>>(irp, iip, lamp, TD);

  // forward scan
  k_scan_pass1<<<NCHUNK * BD / 256, b256, 0, stream>>>(
      lamp, irp, iip, trig, trig + DIM, GrF, GiF, HrF, HiF, 0);
  k_scan_pass2<<<BD / 256, b256, 0, stream>>>(GrF, GiF, HrF, HiF);
  k_scan_pass3<<<NCHUNK * BD / 256, b256, 0, stream>>>(
      lamp, irp, iip, trig, trig + DIM, HrF, HiF, featp, 0, 0);

  // reverse scan (accumulates into feature)
  k_scan_pass1<<<NCHUNK * BD / 256, b256, 0, stream>>>(
      lamp, irp, iip, trig + 2 * DIM, trig + 3 * DIM, GrR, GiR, HrR, HiR, 1);
  k_scan_pass2<<<BD / 256, b256, 0, stream>>>(GrR, GiR, HrR, HiR);
  k_scan_pass3<<<NCHUNK * BD / 256, b256, 0, stream>>>(
      lamp, irp, iip, trig + 2 * DIM, trig + 3 * DIM, HrR, HiR, featp, 1, 1);

  // layernorm * gate -> bf16
  k_ln_gate<<<TOKENS, b256, 0, stream>>>(featp, gatep, lng, lnb, gbf);

  // output GEMM + bias
  k_gemm2_wmma<<<dim3(DIM / 128, TOKENS / 64), b256, 0, stream>>>(
      gbf, wot, bo, (float*)d_out);
}
